// RNNLayer_43508018708740
// MI455X (gfx1250) — compile-verified, matrix operand order checked by reading, not run on hardware
//
#include <hip/hip_runtime.h>
#include <hip/hip_bf16.h>

#define ALPHA_ 0.1f
#define B_ 64
#define T_ 512
#define DIN_ 256
#define H_ 1024
#define DOUT_ 256

typedef __bf16 bf16_t;
typedef __attribute__((ext_vector_type(16))) __bf16 v16bf;
typedef __attribute__((ext_vector_type(8))) float v8f;

union Frag16 { v16bf v; unsigned u[8]; };

// Per-lane K base for 16-bit A/B fragments (CDNA5 WMMA 16x16x32 layout):
// VGPR v<4: K base 0, v>=4: K base 16; lanes 16-31 shifted by 8; 2 packed per dword.
__device__ __forceinline__ int frag_k0(int v, int hi) {
    return ((v & 4) << 2) + (hi << 3) + ((v & 3) << 1);
}

// Generic (flat) LDS pointer -> 32-bit LDS byte offset. Per CDNA5 aperture
// rules the LDS aperture occupies only ADDR[63:32]; ADDR[31:0] is the offset.
__device__ __forceinline__ unsigned lds_off32(const void* p) {
    return (unsigned)(size_t)p;
}

// CDNA5 async global->LDS copy (VGLOBAL encoding, tracked by ASYNCcnt).
__device__ __forceinline__ void async_copy_b64(void* lds_dst, const void* gsrc) {
    asm volatile("global_load_async_to_lds_b64 %0, %1, off"
                 :: "v"(lds_off32(lds_dst)), "v"(gsrc) : "memory");
}
__device__ __forceinline__ void wait_async_all() {
    asm volatile("s_wait_asynccnt 0x0" ::: "memory");
}

// ---------------------------------------------------------------------------
// fp32 -> bf16 weight conversion
// ---------------------------------------------------------------------------
__global__ void cvt_f32_bf16(const float* __restrict__ src, bf16_t* __restrict__ dst, int n) {
    int i = blockIdx.x * blockDim.x + threadIdx.x;
    if (i < n) dst[i] = (bf16_t)src[i];
}

// ---------------------------------------------------------------------------
// C[M,N] = scale * (A[M,K](f32->bf16) @ Bw[N,K]^T (bf16) + bias[N])
// 64x128 block per WG, 8 waves, each wave: 1 M-tile x 4 N-tiles, K-step 32.
// Double-buffered LDS; B tile staged via global_load_async_to_lds_b64,
// A tile via register f32->bf16 conversion. One barrier per K-step.
// ---------------------------------------------------------------------------
__global__ __launch_bounds__(256) void wmma_gemm_bf16(
        const float* __restrict__ A, const bf16_t* __restrict__ Bw,
        const float* __restrict__ bias, float* __restrict__ C,
        int M, int N, int K, float scale) {
    __shared__ bf16_t Asm[2][64][36];    // 72B row stride: 8B-aligned, low-conflict
    __shared__ bf16_t Bsm[2][128][36];

    const int tid  = threadIdx.x;
    const int lane = tid & 31;
    const int wave = tid >> 5;
    const int mbase = blockIdx.y * 64;
    const int nbase = blockIdx.x * 128;
    const int mt  = wave & 3;            // M-tile of this wave
    const int ng  = wave >> 2;           // N-half of this wave
    const int row = lane & 15;
    const int hi  = lane >> 4;

    v8f acc[4];
    #pragma unroll
    for (int j = 0; j < 4; ++j)
        #pragma unroll
        for (int r = 0; r < 8; ++r) acc[j][r] = 0.f;

    float2 areg[4];

    // ---- prologue: stage tile kb=0 into buffer 0 ----
    #pragma unroll
    for (int i = 0; i < 4; ++i) {        // A: 64x32 = 1024 float2 / 256 thr
        int p = tid + i * 256;
        int m = p >> 4, kp = (p & 15) << 1;
        areg[i] = *(const float2*)(A + (size_t)(mbase + m) * K + kp);
    }
    #pragma unroll
    for (int i = 0; i < 4; ++i) {        // B: 128x32 bf16 = 1024 b64 / 256 thr
        int p = tid + i * 256;
        int n = p >> 3, kc = (p & 7) << 2;
        async_copy_b64(&Bsm[0][n][kc], Bw + (size_t)(nbase + n) * K + kc);
    }
    #pragma unroll
    for (int i = 0; i < 4; ++i) {
        int p = tid + i * 256;
        int m = p >> 4, kp = (p & 15) << 1;
        union { unsigned u; bf16_t h[2]; } pk;
        pk.h[0] = (bf16_t)areg[i].x;
        pk.h[1] = (bf16_t)areg[i].y;
        *(unsigned*)&Asm[0][m][kp] = pk.u;
    }
    wait_async_all();
    __syncthreads();

    int cur = 0;
    for (int kb = 0; kb < K; kb += 32) {
        const int nxt = cur ^ 1;
        const bool more = (kb + 32) < K;

        if (more) {
            // issue next tile's global traffic before computing on current
            #pragma unroll
            for (int i = 0; i < 4; ++i) {
                int p = tid + i * 256;
                int m = p >> 4, kp = (p & 15) << 1;
                areg[i] = *(const float2*)(A + (size_t)(mbase + m) * K + kb + 32 + kp);
            }
            #pragma unroll
            for (int i = 0; i < 4; ++i) {
                int p = tid + i * 256;
                int n = p >> 3, kc = (p & 7) << 2;
                async_copy_b64(&Bsm[nxt][n][kc],
                               Bw + (size_t)(nbase + n) * K + kb + 32 + kc);
            }
        }

        // ---- WMMA on current buffer ----
        Frag16 af;
        #pragma unroll
        for (int v = 0; v < 8; ++v)
            af.u[v] = *(const unsigned*)&Asm[cur][mt * 16 + row][frag_k0(v, hi)];
        #pragma unroll
        for (int j = 0; j < 4; ++j) {
            Frag16 bfg;
            int nn = (ng * 4 + j) * 16 + row;
            #pragma unroll
            for (int v = 0; v < 8; ++v)
                bfg.u[v] = *(const unsigned*)&Bsm[cur][nn][frag_k0(v, hi)];
            acc[j] = __builtin_amdgcn_wmma_f32_16x16x32_bf16(
                false, af.v, false, bfg.v, (short)0, acc[j], false, false);
        }

        if (more) {
            // convert + park next A tile
            #pragma unroll
            for (int i = 0; i < 4; ++i) {
                int p = tid + i * 256;
                int m = p >> 4, kp = (p & 15) << 1;
                union { unsigned u; bf16_t h[2]; } pk;
                pk.h[0] = (bf16_t)areg[i].x;
                pk.h[1] = (bf16_t)areg[i].y;
                *(unsigned*)&Asm[nxt][m][kp] = pk.u;
            }
        }
        wait_async_all();
        __syncthreads();
        cur = nxt;
    }

    #pragma unroll
    for (int j = 0; j < 4; ++j) {
        #pragma unroll
        for (int r = 0; r < 8; ++r) {
            int grow = mbase + mt * 16 + (hi ? 8 + r : r);
            int gcol = nbase + (ng * 4 + j) * 16 + row;
            C[(size_t)grow * N + gcol] = scale * (acc[j][r] + bias[gcol]);
        }
    }
}

// ---------------------------------------------------------------------------
// Grid-wide split barrier via monotonic device-scope counter.
// ---------------------------------------------------------------------------
__device__ __forceinline__ void grid_sync(unsigned* cnt, unsigned target) {
    __threadfence();                  // release: publish this WG's stores
    __syncthreads();
    if (threadIdx.x == 0) {
        __hip_atomic_fetch_add(cnt, 1u, __ATOMIC_ACQ_REL, __HIP_MEMORY_SCOPE_AGENT);
        while (__hip_atomic_load(cnt, __ATOMIC_ACQUIRE, __HIP_MEMORY_SCOPE_AGENT) < target)
            __builtin_amdgcn_s_sleep(1);
    }
    __syncthreads();
    __threadfence();                  // acquire: invalidate stale lines
}

// ---------------------------------------------------------------------------
// Persistent CTRNN recurrence for one layer.
//   U[B][T][H] on entry holds alpha*(in_proj + b); on exit holds fr (relu(v)).
//   grid = 64 WGs (one per 16-column slice), 256 threads (8 waves).
//   Wave w owns K-range [w*128, w*128+128): its 4 W_rec B-fragments stay in
//   VGPRs for the whole T loop. v-state lives in registers of waves 0..3.
// ---------------------------------------------------------------------------
__global__ __launch_bounds__(256) void ctrnn_recur(
        float* __restrict__ U, const bf16_t* __restrict__ Wrec,
        bf16_t* __restrict__ frB, unsigned* __restrict__ cnt) {
    __shared__ float red[8][4][32][8];   // [wave][mtile][lane][r] = 32 KB

    const int tid   = threadIdx.x;
    const int lane  = tid & 31;
    const int wave  = tid >> 5;
    const int nbase = blockIdx.x * 16;
    const int col   = lane & 15;
    const int hi    = lane >> 4;
    const int kwave = wave * 128;

    // zero this WG's slice of the fr buffer (t = -1 state)
    for (int i = tid; i < B_ * 16; i += 256) {
        int b = i >> 4;
        frB[(size_t)b * H_ + nbase + (i & 15)] = (bf16_t)0.f;
    }

    // preload W_rec fragments for this wave's K range (resident all T steps)
    Frag16 bfrag[4];
    #pragma unroll
    for (int ks = 0; ks < 4; ++ks) {
        int kb = kwave + ks * 32;
        #pragma unroll
        for (int v = 0; v < 8; ++v)
            bfrag[ks].u[v] = *(const unsigned*)(
                Wrec + (size_t)(nbase + col) * H_ + kb + frag_k0(v, hi));
    }

    float vacc[8];
    #pragma unroll
    for (int r = 0; r < 8; ++r) vacc[r] = 0.f;

    unsigned target = gridDim.x;
    grid_sync(cnt, target);           // fr(-1)=0 visible everywhere

    for (int t = 0; t < T_; ++t) {
        // partial GEMM: fr(t-1)[64xH] @ Wrec^T, this wave's K-range
        #pragma unroll
        for (int m = 0; m < 4; ++m) {
            v8f acc;
            #pragma unroll
            for (int r = 0; r < 8; ++r) acc[r] = 0.f;
            #pragma unroll
            for (int ks = 0; ks < 4; ++ks) {
                int kb = kwave + ks * 32;
                Frag16 af;
                #pragma unroll
                for (int v = 0; v < 8; ++v)
                    af.u[v] = *(const unsigned*)(
                        frB + (size_t)(m * 16 + col) * H_ + kb + frag_k0(v, hi));
                acc = __builtin_amdgcn_wmma_f32_16x16x32_bf16(
                    false, af.v, false, bfrag[ks].v, (short)0, acc, false, false);
            }
            #pragma unroll
            for (int r = 0; r < 8; ++r) red[wave][m][lane][r] = acc[r];
        }
        __syncthreads();

        // K-reduction across 8 waves + leaky/ReLU state update (waves 0..3)
        if (wave < 4) {
            #pragma unroll
            for (int r = 0; r < 8; ++r) {
                float s = 0.f;
                #pragma unroll
                for (int wv = 0; wv < 8; ++wv) s += red[wv][wave][lane][r];
                int brow = wave * 16 + (hi ? 8 + r : r);
                size_t uidx = (size_t)brow * T_ * H_ + (size_t)t * H_ + nbase + col;
                float v = (1.f - ALPHA_) * vacc[r] + ALPHA_ * s + U[uidx];
                vacc[r] = v;
                float fr = v > 0.f ? v : 0.f;
                U[uidx] = fr;                                       // states output
                frB[(size_t)brow * H_ + nbase + col] = (bf16_t)fr;  // next-step A
            }
        }
        target += gridDim.x;
        grid_sync(cnt, target);
    }
}

// ---------------------------------------------------------------------------
extern "C" void kernel_launch(void* const* d_in, const int* in_sizes, int n_in,
                              void* d_out, int out_size, void* d_ws, size_t ws_size,
                              hipStream_t stream) {
    const float* x     = (const float*)d_in[0];
    const float* Win0  = (const float*)d_in[1];
    const float* Wrec0 = (const float*)d_in[2];
    const float* b0    = (const float*)d_in[3];
    const float* Win1  = (const float*)d_in[4];
    const float* Wrec1 = (const float*)d_in[5];
    const float* b1    = (const float*)d_in[6];
    const float* Wout  = (const float*)d_in[7];
    const float* bout  = (const float*)d_in[8];

    float* out = (float*)d_out;                        // [B][T][DOUT]
    float* s0  = out + (size_t)B_ * T_ * DOUT_;        // [B][T][H]
    float* s1  = s0 + (size_t)B_ * T_ * H_;            // [B][T][H]

    char* ws = (char*)d_ws;
    size_t o = 0;
    bf16_t* Win0_b  = (bf16_t*)(ws + o); o += (size_t)H_ * DIN_ * 2;
    bf16_t* Wrec0_b = (bf16_t*)(ws + o); o += (size_t)H_ * H_ * 2;
    bf16_t* Win1_b  = (bf16_t*)(ws + o); o += (size_t)H_ * H_ * 2;
    bf16_t* Wrec1_b = (bf16_t*)(ws + o); o += (size_t)H_ * H_ * 2;
    bf16_t* Wout_b  = (bf16_t*)(ws + o); o += (size_t)DOUT_ * H_ * 2;
    bf16_t* frB     = (bf16_t*)(ws + o); o += (size_t)B_ * H_ * 2;
    unsigned* cnt   = (unsigned*)(ws + o);

    hipMemsetAsync(cnt, 0, 2 * sizeof(unsigned), stream);

    cvt_f32_bf16<<<(H_ * DIN_ + 255) / 256, 256, 0, stream>>>(Win0, Win0_b, H_ * DIN_);
    cvt_f32_bf16<<<(H_ * H_ + 255) / 256, 256, 0, stream>>>(Wrec0, Wrec0_b, H_ * H_);
    cvt_f32_bf16<<<(H_ * H_ + 255) / 256, 256, 0, stream>>>(Win1, Win1_b, H_ * H_);
    cvt_f32_bf16<<<(H_ * H_ + 255) / 256, 256, 0, stream>>>(Wrec1, Wrec1_b, H_ * H_);
    cvt_f32_bf16<<<(DOUT_ * H_ + 255) / 256, 256, 0, stream>>>(Wout, Wout_b, DOUT_ * H_);

    const int M = B_ * T_;

    // U0 = alpha*(x @ Win0^T + b0) -> s0
    wmma_gemm_bf16<<<dim3(H_ / 128, M / 64), 256, 0, stream>>>(
        x, Win0_b, b0, s0, M, H_, DIN_, ALPHA_);
    // layer-0 recurrence, in place on s0
    ctrnn_recur<<<H_ / 16, 256, 0, stream>>>(s0, Wrec0_b, frB, cnt);
    // U1 = alpha*(states0 @ Win1^T + b1) -> s1
    wmma_gemm_bf16<<<dim3(H_ / 128, M / 64), 256, 0, stream>>>(
        s0, Win1_b, b1, s1, M, H_, H_, ALPHA_);
    // layer-1 recurrence, in place on s1
    ctrnn_recur<<<H_ / 16, 256, 0, stream>>>(s1, Wrec1_b, frB, cnt + 1);
    // output = states1 @ Wout^T + bout
    wmma_gemm_bf16<<<dim3(DOUT_ / 128, M / 64), 256, 0, stream>>>(
        s1, Wout_b, bout, out, M, DOUT_, H_, 1.0f);
}
